// MixingBlock_10411000725987
// MI455X (gfx1250) — matrix operand order repaired and checked by reading, not compiled, not run on hardware
//
#include <hip/hip_runtime.h>
#include <hip/hip_bf16.h>
#include <cstdint>
#include <cstddef>

typedef __attribute__((ext_vector_type(16))) _Float16 v16h;
typedef __attribute__((ext_vector_type(8)))  _Float16 h8v;
typedef __attribute__((ext_vector_type(8)))  float    v8f;

#define DEVI __device__ __forceinline__

static constexpr int TOK  = 65536;   // total tokens = B * D * H * W
static constexpr float EPSV = 1e-5f;

enum { EPI_F32 = 0, EPI_QKV = 1, EPI_RESID = 2, EPI_GELU_H = 3, EPI_OUT = 4 };

// ---------------- small device helpers ----------------
DEVI float wred(float v) {
#pragma unroll
  for (int o = 16; o > 0; o >>= 1) v += __shfl_xor(v, o, 32);
  return v;
}

__device__ float block_sum256(float v) {
  __shared__ float sh[8];
  __syncthreads();
  int lane = threadIdx.x & 31, w = threadIdx.x >> 5;
  v = wred(v);
  if (lane == 0) sh[w] = v;
  __syncthreads();
  float r = (lane < 8) ? sh[lane] : 0.0f;
  return wred(r);           // every thread ends with the full block sum
}

DEVI float gelu_f(float x) { return 0.5f * x * (1.0f + erff(x * 0.70710678118654752f)); }
DEVI float sigm(float x)   { return 1.0f / (1.0f + __expf(-x)); }

// window-token index <-> natural (b, l) index.  WS=4, D=16, H=32, W=32
DEVI int nat_from_wt(int t) {
  int wi = t >> 6, nt = t & 63;
  int b = wi >> 8, r = wi & 255;
  int wd = r >> 6, wh = (r >> 3) & 7, ww = r & 7;
  int d = (wd << 2) + (nt >> 4);
  int h = (wh << 2) + ((nt >> 2) & 3);
  int w = (ww << 2) + (nt & 3);
  return (b << 14) + (d << 10) + (h << 5) + w;
}
DEVI int wt_from_nat(int b, int l) {
  int d = l >> 10, h = (l >> 5) & 31, w = l & 31;
  int wi = ((((b << 2) + (d >> 2)) << 3) + (h >> 2)) * 8 + (w >> 2);
  int nt = ((d & 3) << 4) + ((h & 3) << 2) + (w & 3);
  return (wi << 6) + nt;
}

DEVI v8f wmma16x16x32(v16h a, v16h b, v8f c) {
  return __builtin_amdgcn_wmma_f32_16x16x32_f16(false, a, false, b, (short)0, c,
                                                false, false);
}
DEVI v16h cat8(h8v lo, h8v hi) {
  return __builtin_shufflevector(lo, hi, 0, 1, 2, 3, 4, 5, 6, 7,
                                 8, 9, 10, 11, 12, 13, 14, 15);
}

// ---------------- weight cast + transpose:  f32 [K][N] -> f16 [N][K] ----------------
__global__ void k_wcast(const float* __restrict__ in, _Float16* __restrict__ out,
                        int K, int Nc) {
  int idx = blockIdx.x * 256 + threadIdx.x;
  if (idx >= K * Nc) return;
  int k = idx / Nc, nn = idx - k * Nc;
  out[(size_t)nn * K + k] = (_Float16)in[idx];
}

// ---------------- generic WMMA GEMM ----------------
// Each wave computes a 16-row x 64-col strip (4 N-tiles) so one A fragment
// feeds 4 WMMAs: per K-chunk = 2 A B128 loads + 8 B B128 loads -> 4 WMMAs.
// A fragment (16x32 f16): lane group g=lane/16, m=lane&15; halves at
//   k = kc + g*8 + [0..7]  and  k = kc + 16 + g*8 + [0..7]
// B fragment from transposed weights [n][k]: halves at k = kc + g*8 + [0..7] and kc + 16 + g*8 + [0..7]
template<int KDIM, int NOUT, int EPI>
__global__ __launch_bounds__(256) void k_gemm(const _Float16* __restrict__ A,
                                              const _Float16* __restrict__ WT,
                                              const float* __restrict__ bias,
                                              float* __restrict__ fout,
                                              _Float16* __restrict__ hout,
                                              const float* __restrict__ aux) {
  constexpr int NT4 = NOUT / 64;
  int lane = threadIdx.x & 31;
  int tile = blockIdx.x * 8 + (threadIdx.x >> 5);
  int tm = tile / NT4, tn4 = tile - tm * NT4;
  int row0 = tm << 4, col0 = tn4 << 6;
  int g = lane >> 4, mn = lane & 15;

  v8f acc[4];
#pragma unroll
  for (int j = 0; j < 4; ++j) {
    float bv = bias[col0 + j * 16 + mn];
#pragma unroll
    for (int r = 0; r < 8; ++r) acc[j][r] = bv;
  }

  const _Float16* ap = A  + (size_t)(row0 + mn) * KDIM + (g << 3);
  const _Float16* bp = WT + (size_t)(col0 + mn) * KDIM + (g << 3);

  for (int kc = 0; kc < KDIM; kc += 32) {
    h8v alo = *(const h8v*)(ap + kc);
    h8v ahi = *(const h8v*)(ap + kc + 16);
    if (kc + 32 < KDIM) __builtin_prefetch(ap + kc + 32, 0, 0);
    v16h af = cat8(alo, ahi);
#pragma unroll
    for (int j = 0; j < 4; ++j) {
      const _Float16* bj = bp + (size_t)j * 16 * KDIM + kc;
      h8v blo = *(const h8v*)(bj);
      h8v bhi = *(const h8v*)(bj + 16);
      acc[j] = wmma16x16x32(af, cat8(blo, bhi), acc[j]);
    }
  }

#pragma unroll
  for (int j = 0; j < 4; ++j) {
    int o = col0 + j * 16 + mn;
#pragma unroll
    for (int r = 0; r < 8; ++r) {
      int t = row0 + r + (g << 3);
      float vv = acc[j][r];
      if (EPI == EPI_F32) {
        fout[(size_t)t * NOUT + o] = vv;
      } else if (EPI == EPI_QKV) {
        int sec = o >> 7, cc = o & 127;
        size_t base = ((((size_t)(t >> 6)) * 3 + sec) * 8 + (cc >> 4)) * 1024;
        if (sec == 2) {                                      // V: gate + transpose
          vv *= aux[((t >> 14) << 7) + cc];                  // sigmoid(ci) gate
          hout[base + (size_t)(cc & 15) * 64 + (t & 63)] = (_Float16)vv;
        } else {                                             // Q,K: [token][hd]
          hout[base + (size_t)(t & 63) * 16 + (cc & 15)] = (_Float16)vv;
        }
      } else if (EPI == EPI_RESID) {
        size_t nr = (size_t)nat_from_wt(t);                  // window -> natural scatter
        fout[nr * 256 + o] = aux[nr * 256 + o] + vv;         // + shortcut x
      } else if (EPI == EPI_GELU_H) {
        hout[(size_t)t * NOUT + o] = (_Float16)gelu_f(vv);
      } else {                                               // EPI_OUT
        fout[(size_t)t * 256 + o] = aux[(size_t)t * 256 + o] + vv;   // + x1
      }
    }
  }
}

// ---------------- layer norms ----------------
__global__ __launch_bounds__(256) void k_norm1(const float* __restrict__ x,
                                               const float* __restrict__ gw,
                                               const float* __restrict__ bw,
                                               _Float16* __restrict__ xw) {
  int warp = threadIdx.x >> 5, lane = threadIdx.x & 31;
  int tok = blockIdx.x * 8 + warp;                 // natural token
  const float* rp = x + (size_t)tok * 256;
  float v[8], s = 0.f, ss = 0.f;
#pragma unroll
  for (int i = 0; i < 8; ++i) { float t = rp[lane + i * 32]; v[i] = t; s += t; ss += t * t; }
  s = wred(s); ss = wred(ss);
  float mean = s * (1.f / 256.f);
  float rstd = rsqrtf(ss * (1.f / 256.f) - mean * mean + EPSV);
  int wt = wt_from_nat(tok >> 14, tok & 16383);
  _Float16* op = xw + (size_t)wt * 256;
#pragma unroll
  for (int i = 0; i < 8; ++i) {
    int c = lane + i * 32;
    op[c] = (_Float16)((v[i] - mean) * rstd * gw[c] + bw[c]);
  }
}

__global__ __launch_bounds__(256) void k_ln128(const float* __restrict__ io,
                                               const float* __restrict__ gw,
                                               const float* __restrict__ bw,
                                               _Float16* __restrict__ hout,
                                               int hstride) {
  int warp = threadIdx.x >> 5, lane = threadIdx.x & 31;
  size_t t = (size_t)blockIdx.x * 8 + warp;
  const float* rp = io + t * 128;
  float v[4], s = 0.f, ss = 0.f;
#pragma unroll
  for (int i = 0; i < 4; ++i) { float u = rp[lane + i * 32]; v[i] = u; s += u; ss += u * u; }
  s = wred(s); ss = wred(ss);
  float mean = s * (1.f / 128.f);
  float rstd = rsqrtf(ss * (1.f / 128.f) - mean * mean + EPSV);
#pragma unroll
  for (int i = 0; i < 4; ++i) {
    int c = lane + i * 32;
    hout[t * hstride + c] = (_Float16)((v[i] - mean) * rstd * gw[c] + bw[c]);
  }
}

__global__ __launch_bounds__(256) void k_ln_pcn(const float* __restrict__ gin,
                                                const float* __restrict__ gw,
                                                const float* __restrict__ bw,
                                                float* __restrict__ sp) {
  int warp = threadIdx.x >> 5, lane = threadIdx.x & 31;
  int t = blockIdx.x * 8 + warp;                   // window token
  const float* rp = gin + (size_t)t * 256;
  float v[8], s = 0.f, ss = 0.f;
#pragma unroll
  for (int i = 0; i < 8; ++i) { float u = rp[lane + i * 32]; v[i] = u; s += u; ss += u * u; }
  s = wred(s); ss = wred(ss);
  float mean = s * (1.f / 256.f);
  float rstd = rsqrtf(ss * (1.f / 256.f) - mean * mean + EPSV);
  int natl = nat_from_wt(t); int b = natl >> 14; int l = natl & 16383;
#pragma unroll
  for (int i = 0; i < 8; ++i) {
    int c = lane + i * 32;
    sp[(((size_t)((b << 8) + c)) << 14) + l] = (v[i] - mean) * rstd * gw[c] + bw[c];
  }
}

__global__ __launch_bounds__(256) void k_ln256_h(const float* __restrict__ in,
                                                 const float* __restrict__ gw,
                                                 const float* __restrict__ bw,
                                                 _Float16* __restrict__ hout) {
  int warp = threadIdx.x >> 5, lane = threadIdx.x & 31;
  size_t t = (size_t)blockIdx.x * 8 + warp;
  const float* rp = in + t * 256;
  float v[8], s = 0.f, ss = 0.f;
#pragma unroll
  for (int i = 0; i < 8; ++i) { float u = rp[lane + i * 32]; v[i] = u; s += u; ss += u * u; }
  s = wred(s); ss = wred(ss);
  float mean = s * (1.f / 256.f);
  float rstd = rsqrtf(ss * (1.f / 256.f) - mean * mean + EPSV);
#pragma unroll
  for (int i = 0; i < 8; ++i) {
    int c = lane + i * 32;
    hout[t * 256 + c] = (_Float16)((v[i] - mean) * rstd * gw[c] + bw[c]);
  }
}

// ---------------- depthwise 3x3x3 conv (NCDHW, SAME, groups=C) ----------------
__global__ __launch_bounds__(256) void k_dwconv(const float* __restrict__ in,
                                                const float* __restrict__ w,
                                                const float* __restrict__ bias,
                                                float* __restrict__ out) {
  size_t i = (size_t)blockIdx.x * 256 + threadIdx.x;   // over 4*256*16*32*32 = 2^24
  int xw_ = i & 31, y = (i >> 5) & 31, z = (i >> 10) & 15, c = (i >> 14) & 255;
  const float* wp = w + c * 27;
  const float* base = in + ((i >> 14) << 14);
  float s = 0.f;
#pragma unroll
  for (int dz = -1; dz <= 1; ++dz) {
    int zz = z + dz; if (zz < 0 || zz > 15) continue;
#pragma unroll
    for (int dy = -1; dy <= 1; ++dy) {
      int yy = y + dy; if ((unsigned)yy > 31u) continue;
#pragma unroll
      for (int dx = -1; dx <= 1; ++dx) {
        int xx = xw_ + dx; if ((unsigned)xx > 31u) continue;
        s += base[(zz << 10) + (yy << 5) + xx] * wp[(dz + 1) * 9 + (dy + 1) * 3 + (dx + 1)];
      }
    }
  }
  out[i] = s + bias[c];
}

// ---------------- instance-norm stats / reductions ----------------
__global__ __launch_bounds__(256) void k_stats_contig(const float* __restrict__ in,
                                                      float* __restrict__ stats) {
  const float* p = in + ((size_t)blockIdx.x << 14);
  float s = 0.f, ss = 0.f;
  for (int i = threadIdx.x; i < 16384; i += 256) { float v = p[i]; s += v; ss += v * v; }
  s = block_sum256(s); ss = block_sum256(ss);
  if (threadIdx.x == 0) {
    float m = s * (1.f / 16384.f);
    stats[2 * blockIdx.x] = m;
    stats[2 * blockIdx.x + 1] = rsqrtf(ss * (1.f / 16384.f) - m * m + EPSV);
  }
}

__global__ __launch_bounds__(256) void k_stats_strided(const float* __restrict__ in,
                                                       const float* __restrict__ gatep,
                                                       float* __restrict__ stats, int CH) {
  int grp = blockIdx.x; int b = grp / CH, ch = grp - b * CH;
  const float* base = in + ((size_t)b * 16384) * CH + ch;
  const float* gp = gatep ? gatep + (size_t)b * 16384 : nullptr;
  float s = 0.f, ss = 0.f;
  for (int i = threadIdx.x; i < 16384; i += 256) {
    float v = base[(size_t)i * CH];
    if (gp) v *= gp[i];
    s += v; ss += v * v;
  }
  s = block_sum256(s); ss = block_sum256(ss);
  if (threadIdx.x == 0) {
    float m = s * (1.f / 16384.f);
    stats[2 * grp] = m;
    stats[2 * grp + 1] = rsqrtf(ss * (1.f / 16384.f) - m * m + EPSV);
  }
}

// normalize + GELU, token-major f16 copy (window order)
__global__ __launch_bounds__(256) void k_gelu_tok(const float* __restrict__ sp2,
                                                  const float* __restrict__ st,
                                                  _Float16* __restrict__ xcg) {
  size_t idx = (size_t)blockIdx.x * 256 + threadIdx.x;
  int t = (int)(idx >> 8), c = (int)(idx & 255);
  int natl = nat_from_wt(t); int b = natl >> 14, l = natl & 16383;
  int grp = (b << 8) + c;
  float v = sp2[((size_t)grp << 14) + l];
  xcg[idx] = (_Float16)gelu_f((v - st[2 * grp]) * st[2 * grp + 1]);
}

__global__ __launch_bounds__(256) void k_gap(const float* __restrict__ sp2,
                                             const float* __restrict__ st,
                                             float* __restrict__ gap) {
  int grp = blockIdx.x;
  float m = st[2 * grp], r = st[2 * grp + 1];
  const float* p = sp2 + ((size_t)grp << 14);
  float s = 0.f;
  for (int i = threadIdx.x; i < 16384; i += 256) s += gelu_f((p[i] - m) * r);
  s = block_sum256(s);
  if (threadIdx.x == 0) gap[grp] = s * (1.f / 16384.f);
}

// channel interaction MLP -> sigmoid gate (4 x 128)
__global__ void k_ci(const float* __restrict__ gap, const float* __restrict__ w1,
                     const float* __restrict__ b1, const float* __restrict__ w2,
                     const float* __restrict__ b2, float* __restrict__ gate) {
  __shared__ float hsh[32];
  int tid = threadIdx.x;
  for (int b = 0; b < 4; ++b) {
    if (tid < 32) {
      float s = b1[tid];
      for (int c = 0; c < 256; ++c) s += gap[b * 256 + c] * w1[c * 32 + tid];
      hsh[tid] = gelu_f(s);
    }
    __syncthreads();
    float s2 = b2[tid];
    for (int j = 0; j < 32; ++j) s2 += hsh[j] * w2[j * 128 + tid];
    gate[b * 128 + tid] = sigm(s2);
    __syncthreads();
  }
}

// ---------------- windowed attention (one wave per window-head) ----------------
// qkv layout per (window, head): Q,K = [token(64)][hd(16)];  V = [hd(16)][token(64)]
__global__ __launch_bounds__(32) void k_attn(const _Float16* __restrict__ qkv,
                                             const float* __restrict__ rpb,
                                             float* __restrict__ out) {
  __shared__ float    S[64 * 64];
  __shared__ _Float16 P[64 * 64];
  int lane = threadIdx.x;
  int wi = blockIdx.x >> 3, head = blockIdx.x & 7;
  const _Float16* qp = qkv + (((size_t)wi * 3 + 0) * 8 + head) * 1024;
  const _Float16* kp = qp + 8 * 1024;
  const _Float16* vp = qp + 16 * 1024;     // transposed [hd][token]
  int g = lane >> 4, mn = lane & 15;

  // lane-uniform 0/1 mask for the upper-K half (hd=16 zero-padded to K=32)
  _Float16 kmask = (g == 0) ? (_Float16)1.f : (_Float16)0.f;
  h8v kms;
#pragma unroll
  for (int i = 0; i < 8; ++i) kms[i] = kmask;
  h8v zero8;
#pragma unroll
  for (int i = 0; i < 8; ++i) zero8[i] = (_Float16)0.f;

  // S = (q * 0.25) @ k^T
#pragma unroll
  for (int ti = 0; ti < 4; ++ti) {
    h8v qlo = *(const h8v*)(qp + (ti * 16 + mn) * 16 + g * 8);
#pragma unroll
    for (int i = 0; i < 8; ++i) qlo[i] = qlo[i] * (_Float16)0.25f;
    v16h af = cat8(qlo, zero8);            // K in [0,16): data, [16,32): zero
#pragma unroll
    for (int tj = 0; tj < 4; ++tj) {
      const _Float16* kr = kp + (tj * 16 + mn) * 16;
      h8v klo = *(const h8v*)(kr + g * 8); // always in-bounds
      h8v khi = *(const h8v*)(kr + 8);
      khi = khi * kms;
      v8f cf;
#pragma unroll
      for (int r = 0; r < 8; ++r) cf[r] = 0.f;
      cf = wmma16x16x32(af, cat8(klo, khi), cf);
#pragma unroll
      for (int r = 0; r < 8; ++r)
        S[(ti * 16 + r + (g << 3)) * 64 + tj * 16 + mn] = cf[r];
    }
  }
  __syncthreads();

  // softmax rows with relative-position bias from intra-window coords
  for (int rr = 0; rr < 2; ++rr) {
    int row = lane + rr * 32;
    int rd = row >> 4, rh = (row >> 2) & 3, rw = row & 3;
    float mx = -3.0e38f;
    for (int c0 = 0; c0 < 64; ++c0) {
      int rel = (rd - (c0 >> 4) + 3) * 49 + (rh - ((c0 >> 2) & 3) + 3) * 7 + (rw - (c0 & 3) + 3);
      float v = S[row * 64 + c0] + rpb[rel * 8 + head];
      S[row * 64 + c0] = v;
      mx = fmaxf(mx, v);
    }
    float sum = 0.f;
    for (int c0 = 0; c0 < 64; ++c0) {
      float e = __expf(S[row * 64 + c0] - mx);
      S[row * 64 + c0] = e; sum += e;
    }
    float inv = 1.0f / sum;
    for (int c0 = 0; c0 < 64; ++c0)
      P[row * 64 + c0] = (_Float16)(S[row * 64 + c0] * inv);
  }
  __syncthreads();

  // O = P @ V  (64x64 @ 64x16), V already transposed -> contiguous B fragments
#pragma unroll
  for (int ti = 0; ti < 4; ++ti) {
    const _Float16* pr = &P[(ti * 16 + mn) * 64];
    v8f cf;
#pragma unroll
    for (int r = 0; r < 8; ++r) cf[r] = 0.f;
#pragma unroll
    for (int kc = 0; kc < 64; kc += 32) {
      h8v plo = *(const h8v*)(pr + kc + g * 8);
      h8v phi = *(const h8v*)(pr + kc + 16 + g * 8);
      const _Float16* vr = vp + mn * 64 + kc + g * 8;
      h8v vlo = *(const h8v*)(vr);
      h8v vhi = *(const h8v*)(vr + 16);
      cf = wmma16x16x32(cat8(plo, phi), cat8(vlo, vhi), cf);
    }
#pragma unroll
    for (int r = 0; r < 8; ++r)
      out[((size_t)wi * 64 + ti * 16 + r + (g << 3)) * 128 + head * 16 + mn] = cf[r];
  }
}

// ---------------- spatial interaction path ----------------
__global__ __launch_bounds__(256) void k_si1(const float* __restrict__ xa,
                                             const float* __restrict__ w1,
                                             const float* __restrict__ b1,
                                             float* __restrict__ s1) {
  size_t idx = (size_t)blockIdx.x * 256 + threadIdx.x;   // T*16
  int t = (int)(idx >> 4), ch = (int)(idx & 15);
  const float* rp = xa + (size_t)t * 128;
  float s = b1[ch];
  for (int c = 0; c < 128; ++c) s += rp[c] * w1[c * 16 + ch];
  s1[idx] = s;
}

__global__ __launch_bounds__(256) void k_si2(const float* __restrict__ s1,
                                             const float* __restrict__ st,
                                             const float* __restrict__ w2,
                                             const float* __restrict__ b2,
                                             float* __restrict__ sg) {
  int t = blockIdx.x * 256 + threadIdx.x;
  int b = t >> 14;
  float acc = b2[0];
#pragma unroll
  for (int ch = 0; ch < 16; ++ch) {
    int grp = (b << 4) + ch;
    float v = gelu_f((s1[(size_t)t * 16 + ch] - st[2 * grp]) * st[2 * grp + 1]);
    acc += v * w2[ch];
  }
  sg[t] = sigm(acc);
}

__global__ __launch_bounds__(256) void k_xcfinal(const float* __restrict__ xcp,
                                                 const float* __restrict__ sg,
                                                 const float* __restrict__ st,
                                                 _Float16* __restrict__ cat) {
  size_t idx = (size_t)blockIdx.x * 256 + threadIdx.x;   // T*128
  int t = (int)(idx >> 7), c = (int)(idx & 127);
  int grp = ((t >> 14) << 7) + c;
  float v = sg[t] * xcp[idx];
  cat[(size_t)t * 256 + 128 + c] = (_Float16)((v - st[2 * grp]) * st[2 * grp + 1]);
}

// ---------------- host side ----------------
extern "C" void kernel_launch(void* const* d_in, const int* in_sizes, int n_in,
                              void* d_out, int out_size, void* d_ws, size_t ws_size,
                              hipStream_t stream) {
  (void)in_sizes; (void)n_in; (void)out_size; (void)ws_size;
  const float* x     = (const float*)d_in[0];
  const float* n1g   = (const float*)d_in[1];
  const float* n1b   = (const float*)d_in[2];
  const float* rpb   = (const float*)d_in[3];
  const float* paw   = (const float*)d_in[4];
  const float* pab   = (const float*)d_in[5];
  const float* pang  = (const float*)d_in[6];
  const float* panb  = (const float*)d_in[7];
  const float* pcw   = (const float*)d_in[8];
  const float* pcb   = (const float*)d_in[9];
  const float* pcng  = (const float*)d_in[10];
  const float* pcnb  = (const float*)d_in[11];
  const float* dww   = (const float*)d_in[12];
  const float* dwb   = (const float*)d_in[13];
  const float* ciw1  = (const float*)d_in[14];
  const float* cib1  = (const float*)d_in[15];
  const float* ciw2  = (const float*)d_in[16];
  const float* cib2  = (const float*)d_in[17];
  const float* pjcw  = (const float*)d_in[18];
  const float* pjcb  = (const float*)d_in[19];
  const float* qkvw  = (const float*)d_in[20];
  const float* qkvb  = (const float*)d_in[21];
  const float* siw1  = (const float*)d_in[22];
  const float* sib1  = (const float*)d_in[23];
  const float* siw2  = (const float*)d_in[24];
  const float* sib2  = (const float*)d_in[25];
  const float* ang   = (const float*)d_in[26];
  const float* anb   = (const float*)d_in[27];
  const float* prw   = (const float*)d_in[28];
  const float* prb   = (const float*)d_in[29];
  const float* n2g   = (const float*)d_in[30];
  const float* n2b   = (const float*)d_in[31];
  const float* f1w   = (const float*)d_in[32];
  const float* f1b   = (const float*)d_in[33];
  const float* f2w   = (const float*)d_in[34];
  const float* f2b   = (const float*)d_in[35];
  float* out = (float*)d_out;

  char* wsp = (char*)d_ws;
  size_t off = 0;
  auto alloc = [&](size_t bytes) -> void* {
    void* p = wsp + off;
    off += (bytes + 255) & ~(size_t)255;
    return p;
  };
  _Float16* paT   = (_Float16*)alloc(2ull * 256 * 128);
  _Float16* pcT   = (_Float16*)alloc(2ull * 256 * 256);
  _Float16* qkT   = (_Float16*)alloc(2ull * 128 * 384);
  _Float16* pjT   = (_Float16*)alloc(2ull * 256 * 128);
  _Float16* prT   = (_Float16*)alloc(2ull * 256 * 256);
  _Float16* f1T   = (_Float16*)alloc(2ull * 256 * 1024);
  _Float16* f2T   = (_Float16*)alloc(2ull * 1024 * 256);
  _Float16* xw_h  = (_Float16*)alloc(2ull * TOK * 256);
  float*    xa    = (float*)   alloc(4ull * TOK * 128);
  _Float16* xa_h  = (_Float16*)alloc(2ull * TOK * 128);
  float*    g1    = (float*)   alloc(4ull * TOK * 256);  // proj_cnn out; reused as x1
  float*    sp1   = (float*)   alloc(4ull * TOK * 256);  // spatial; reused (w/ sp2) as h2_h
  float*    sp2   = (float*)   alloc(4ull * TOK * 256);
  float*    st1   = (float*)   alloc(4ull * 2048);
  _Float16* xcg_h = (_Float16*)alloc(2ull * TOK * 256);
  float*    gap   = (float*)   alloc(4ull * 1024);
  float*    gate  = (float*)   alloc(4ull * 512);
  float*    xcp   = (float*)   alloc(4ull * TOK * 128);
  _Float16* qkv_h = (_Float16*)alloc(2ull * TOK * 384);
  float*    xat   = (float*)   alloc(4ull * TOK * 128);
  float*    s1    = (float*)   alloc(4ull * TOK * 16);
  float*    st2   = (float*)   alloc(4ull * 128);
  float*    sgate = (float*)   alloc(4ull * TOK);
  float*    st3   = (float*)   alloc(4ull * 1024);
  _Float16* cat_h = (_Float16*)alloc(2ull * TOK * 256);
  float*    x1    = g1;                 // reuse (g1 consumed before proj)
  _Float16* h1_h  = xw_h;               // reuse (xw_h consumed before norm2)
  _Float16* h2_h  = (_Float16*)sp1;     // reuse sp1+sp2 contiguously (128 MB)

  // weight casts (f32 [K][N] -> f16 [N][K])
  k_wcast<<<128,  256, 0, stream>>>(paw,  paT, 256, 128);
  k_wcast<<<256,  256, 0, stream>>>(pcw,  pcT, 256, 256);
  k_wcast<<<192,  256, 0, stream>>>(qkvw, qkT, 128, 384);
  k_wcast<<<128,  256, 0, stream>>>(pjcw, pjT, 256, 128);
  k_wcast<<<256,  256, 0, stream>>>(prw,  prT, 256, 256);
  k_wcast<<<1024, 256, 0, stream>>>(f1w,  f1T, 256, 1024);
  k_wcast<<<1024, 256, 0, stream>>>(f2w,  f2T, 1024, 256);

  // norm1 + window partition
  k_norm1<<<8192, 256, 0, stream>>>(x, n1g, n1b, xw_h);

  // attention projection + pan LN
  k_gemm<256, 128, EPI_F32><<<1024, 256, 0, stream>>>(xw_h, paT, pab, xa, nullptr, nullptr);
  k_ln128<<<8192, 256, 0, stream>>>(xa, pang, panb, xa_h, 128);

  // conv projection + pcn LN -> spatial
  k_gemm<256, 256, EPI_F32><<<2048, 256, 0, stream>>>(xw_h, pcT, pcb, g1, nullptr, nullptr);
  k_ln_pcn<<<8192, 256, 0, stream>>>(g1, pcng, pcnb, sp1);

  // depthwise conv -> instance norm -> gelu; GAP for channel interaction
  k_dwconv<<<65536, 256, 0, stream>>>(sp1, dww, dwb, sp2);
  k_stats_contig<<<1024, 256, 0, stream>>>(sp2, st1);
  k_gelu_tok<<<65536, 256, 0, stream>>>(sp2, st1, xcg_h);
  k_gap<<<1024, 256, 0, stream>>>(sp2, st1, gap);
  k_ci<<<1, 128, 0, stream>>>(gap, ciw1, cib1, ciw2, cib2, gate);

  // projc + qkv (with V gating + V transpose) + attention
  k_gemm<256, 128, EPI_F32><<<1024, 256, 0, stream>>>(xcg_h, pjT, pjcb, xcp, nullptr, nullptr);
  k_gemm<128, 384, EPI_QKV><<<3072, 256, 0, stream>>>(xa_h, qkT, qkvb, nullptr, qkv_h, gate);
  k_attn<<<8192, 32, 0, stream>>>(qkv_h, rpb, xat);

  // spatial interaction: matvec -> inorm -> gelu -> proj -> sigmoid
  k_si1<<<4096, 256, 0, stream>>>(xat, siw1, sib1, s1);
  k_stats_strided<<<64, 256, 0, stream>>>(s1, nullptr, st2, 16);
  k_si2<<<256, 256, 0, stream>>>(s1, st2, siw2, sib2, sgate);

  // gated conv branch instance-norm + concat
  k_stats_strided<<<512, 256, 0, stream>>>(xcp, sgate, st3, 128);
  k_ln128<<<8192, 256, 0, stream>>>(xat, ang, anb, cat_h, 256);
  k_xcfinal<<<32768, 256, 0, stream>>>(xcp, sgate, st3, cat_h);

  // output projection + residual (scatter to natural order), then MLP
  k_gemm<256, 256, EPI_RESID><<<2048, 256, 0, stream>>>(cat_h, prT, prb, x1, nullptr, x);
  k_ln256_h<<<8192, 256, 0, stream>>>(x1, n2g, n2b, h1_h);
  k_gemm<256, 1024, EPI_GELU_H><<<8192, 256, 0, stream>>>(h1_h, f1T, f1b, nullptr, h2_h, nullptr);
  k_gemm<1024, 256, EPI_OUT><<<2048, 256, 0, stream>>>(h2_h, f2T, f2b, out, nullptr, x1);
}
